// DeltaResBlock_9560597201201
// MI455X (gfx1250) — compile-verified, matrix operand order checked by reading, not run on hardware
//
#include <hip/hip_runtime.h>
#include <math.h>

// ---------------------------------------------------------------------------
// DeltaResBlock on MI455X (gfx1250, wave32, WMMA)
//   B=1024, D=512, V=256, NODE=1024, GATE=128
//   Pass 1: pool over x (1 HBM read of x)
//   Small GEMMs via v_wmma_f32_16x16x32_bf16
//   Pass 2: fused proj + rank-1 update, x staged bf16 in 256KB LDS
// ---------------------------------------------------------------------------

#define B_    1024
#define D_    512
#define V_    256
#define NODE_ 1024
#define GATE_ 128

typedef __attribute__((ext_vector_type(16))) __bf16 v16bf;
typedef __attribute__((ext_vector_type(8)))  float  v8f;

__device__ __forceinline__ unsigned short f2bf(float f) {
  unsigned int u = __float_as_uint(f);
  u += 0x7FFFu + ((u >> 16) & 1u);            // round-to-nearest-even
  return (unsigned short)(u >> 16);
}
__device__ __forceinline__ float bf2f(unsigned short h) {
  return __uint_as_float(((unsigned int)h) << 16);
}

// ---------------------------------------------------------------------------
// Weight convert: fp32 [K,N] row-major -> bf16 [N,K] row-major (i.e. K-major
// per output column) so WMMA B fragments load as two b128 per lane.
// ---------------------------------------------------------------------------
__global__ void convert_transpose(const float* __restrict__ W,
                                  unsigned short* __restrict__ Wt,
                                  int K, int N) {
  int idx = blockIdx.x * blockDim.x + threadIdx.x;
  if (idx >= K * N) return;
  int k = idx / N, n = idx % N;
  Wt[(size_t)n * K + k] = f2bf(W[idx]);
}

// ---------------------------------------------------------------------------
// Pooling: one workgroup per batch b. Thread tid owns column v = tid.
// Produces pool_d[b,:] (row means) and pool_v[b,:] (column means), bf16.
// ---------------------------------------------------------------------------
__global__ void pool_kernel(const float* __restrict__ x,
                            unsigned short* __restrict__ pool_d,
                            unsigned short* __restrict__ pool_v) {
  __shared__ float rowAcc[D_];
  const int b = blockIdx.x, tid = threadIdx.x, lane = tid & 31;
  for (int d = tid; d < D_; d += blockDim.x) rowAcc[d] = 0.f;
  __syncthreads();

  const float* xb = x + (size_t)b * D_ * V_;
  float col = 0.f;
  for (int d = 0; d < D_; ++d) {
    float val = bf2f(f2bf(xb[(size_t)d * V_ + tid]));   // cast to bf16 first
    col += val;
    float s = val;
    #pragma unroll
    for (int off = 16; off; off >>= 1) s += __shfl_xor(s, off, 32);
    if (lane == 0) atomicAdd(&rowAcc[d], s);            // ds_add_f32
  }
  __syncthreads();
  for (int d = tid; d < D_; d += blockDim.x)
    pool_d[(size_t)b * D_ + d] = f2bf(rowAcc[d] * (1.f / (float)V_));
  pool_v[(size_t)b * V_ + tid] = f2bf(col * (1.f / (float)D_));
}

// ---------------------------------------------------------------------------
// WMMA GEMM: C[M,N] = act(scale * (A[M,K] @ W[K,N] + bias[N]))
//   A:  bf16 row-major  M x K
//   Wt: bf16 row-major  N x K  (= W column-major)
//   One wave per 16x16 output tile; K-loop step 32 (v_wmma_f32_16x16x32_bf16).
// Fragment packing (lane L, g=L>>4, r=L&15):
//   element e -> K index: e<8 ? e+8g : 16+(e-8)+8g   (A table; B mirrored)
//   = two 16-byte contiguous chunks at +8g and +16+8g.
// ACT: 0=none, 1=scale-then-relu (BN inference), 2=tanh
// ---------------------------------------------------------------------------
template <int ACT>
__global__ void wmma_gemm(const unsigned short* __restrict__ A,
                          const unsigned short* __restrict__ Wt,
                          const float* __restrict__ bias,
                          unsigned short* __restrict__ C,
                          int M, int N, int K, float scale) {
  const int wave = (blockIdx.x * blockDim.x + threadIdx.x) >> 5;
  const int lane = threadIdx.x & 31;
  const int tilesN = N >> 4;
  const int tm = (wave / tilesN) << 4;
  const int tn = (wave % tilesN) << 4;
  if (tm >= M) return;                         // wave-uniform; EXEC stays all-1
  const int g = lane >> 4;                     // lane half
  const int r = lane & 15;                     // A-row / B-col within tile

  union Frag { uint4 q[2]; v16bf v; };
  v8f acc = {};

  const unsigned short* Arow = A  + (size_t)(tm + r) * K;
  const unsigned short* Brow = Wt + (size_t)(tn + r) * K;

  for (int k0 = 0; k0 < K; k0 += 32) {
    Frag a, bfr;
    a.q[0]   = *(const uint4*)(Arow + k0 + 8 * g);        // K = 8g..8g+7
    a.q[1]   = *(const uint4*)(Arow + k0 + 16 + 8 * g);   // K = 16+8g..23+8g
    bfr.q[0] = *(const uint4*)(Brow + k0 + 8 * g);
    bfr.q[1] = *(const uint4*)(Brow + k0 + 16 + 8 * g);
    acc = __builtin_amdgcn_wmma_f32_16x16x32_bf16(
        false, a.v, false, bfr.v, (short)0, acc, false, false);
  }

  const float bs = bias[tn + r];               // bias per output column
  unsigned short* Crow = C + (size_t)tm * N + (tn + r);
  #pragma unroll
  for (int j = 0; j < 8; ++j) {                // lane holds col r, rows g*8+j
    float xv = acc[j] + bs;
    if (ACT == 1) { xv *= scale; xv = xv > 0.f ? xv : 0.f; }
    else if (ACT == 2) { xv = tanhf(xv); }
    Crow[(size_t)(g * 8 + j) * N] = f2bf(xv);
  }
}

// ---------------------------------------------------------------------------
// Row L2-normalize of k [B, D] bf16 in place: k /= (||k|| + 1e-6)
// ---------------------------------------------------------------------------
__global__ void normalize_k(unsigned short* __restrict__ kmat) {
  __shared__ float red[8];
  __shared__ float invn;
  const int b = blockIdx.x, tid = threadIdx.x;
  float s = 0.f;
  for (int d = tid; d < D_; d += blockDim.x) {
    float v = bf2f(kmat[(size_t)b * D_ + d]);
    s += v * v;
  }
  #pragma unroll
  for (int off = 16; off; off >>= 1) s += __shfl_xor(s, off, 32);
  if ((tid & 31) == 0) red[tid >> 5] = s;
  __syncthreads();
  if (tid == 0) {
    float t = 0.f;
    #pragma unroll
    for (int i = 0; i < 8; ++i) t += red[i];
    invn = 1.f / (sqrtf(t) + 1e-6f);
  }
  __syncthreads();
  const float inv = invn;
  for (int d = tid; d < D_; d += blockDim.x)
    kmat[(size_t)b * D_ + d] = f2bf(bf2f(kmat[(size_t)b * D_ + d]) * inv);
}

// ---------------------------------------------------------------------------
// beta[b] = 2*sigmoid( G[b,:] . bf16(W_b2) + b_b2 ), one wave per b
// ---------------------------------------------------------------------------
__global__ void beta_kernel(const unsigned short* __restrict__ G,
                            const float* __restrict__ Wb2,
                            const float* __restrict__ bb2,
                            float* __restrict__ beta) {
  const int wave = (blockIdx.x * blockDim.x + threadIdx.x) >> 5;
  const int lane = threadIdx.x & 31;
  if (wave >= B_) return;
  float s = 0.f;
  for (int i = lane; i < GATE_; i += 32)
    s += bf2f(G[(size_t)wave * GATE_ + i]) * bf2f(f2bf(Wb2[i]));
  #pragma unroll
  for (int off = 16; off; off >>= 1) s += __shfl_xor(s, off, 32);
  if (lane == 0) beta[wave] = 2.f / (1.f + expf(-(s + bb2[0])));
}

// ---------------------------------------------------------------------------
// Fused delta update. One workgroup per b, 256 threads (thread = v-column).
// Sweep 1: read x[b] fp32 from HBM, stage bf16 into LDS (256 KB),
//          accumulate proj[v] = sum_d k[d]*x[d,v].
// Sweep 2: out = relu(x + beta*k*(v - proj)) read from LDS, write bf16.
// Dynamic LDS: D_*V_*2 (x tile) + D_*2 (k row) = 263168 B (<= 320KB/WGP).
// ---------------------------------------------------------------------------
__global__ void fuse_kernel(const float* __restrict__ x,
                            const unsigned short* __restrict__ kmat,
                            const unsigned short* __restrict__ vmat,
                            const float* __restrict__ beta,
                            unsigned short* __restrict__ out) {
  extern __shared__ char smem[];
  unsigned short* xs = (unsigned short*)smem;                 // [D_ * V_]
  unsigned short* kR = (unsigned short*)(smem + D_ * V_ * 2); // [D_]

  const int b = blockIdx.x, tid = threadIdx.x;
  for (int d = tid; d < D_; d += blockDim.x)
    kR[d] = kmat[(size_t)b * D_ + d];
  __syncthreads();

  const float* xb = x + (size_t)b * D_ * V_;
  float proj = 0.f;
  for (int d = 0; d < D_; ++d) {
    unsigned short h = f2bf(xb[(size_t)d * V_ + tid]);
    xs[d * V_ + tid] = h;
    proj += bf2f(kR[d]) * bf2f(h);
  }

  const float vv   = bf2f(vmat[(size_t)b * V_ + tid]);
  const float bet  = beta[b];
  const float diff = vv - proj;

  unsigned short* ob = out + (size_t)b * D_ * V_;
  for (int d = 0; d < D_; ++d) {
    float o = bf2f(xs[d * V_ + tid]) + bet * bf2f(kR[d]) * diff;
    o = o > 0.f ? o : 0.f;
    ob[(size_t)d * V_ + tid] = f2bf(o);
  }
}

// ---------------------------------------------------------------------------
extern "C" void kernel_launch(void* const* d_in, const int* in_sizes, int n_in,
                              void* d_out, int out_size, void* d_ws, size_t ws_size,
                              hipStream_t stream) {
  (void)in_sizes; (void)n_in; (void)out_size; (void)ws_size;

  const float* x0  = (const float*)d_in[0];
  const float* Wk1 = (const float*)d_in[1];
  const float* bk1 = (const float*)d_in[2];
  const float* Wk2 = (const float*)d_in[3];
  const float* bk2 = (const float*)d_in[4];
  const float* Wv1 = (const float*)d_in[5];
  const float* bv1 = (const float*)d_in[6];
  const float* Wv2 = (const float*)d_in[7];
  const float* bv2 = (const float*)d_in[8];
  const float* Wb1 = (const float*)d_in[9];
  const float* bb1 = (const float*)d_in[10];
  const float* Wb2 = (const float*)d_in[11];
  const float* bb2 = (const float*)d_in[12];
  unsigned short* out = (unsigned short*)d_out;

  // ---- carve workspace (256B aligned slabs) ----
  char* ws = (char*)d_ws;
  size_t off = 0;
  auto carve = [&](size_t bytes) {
    void* p = ws + off;
    off += (bytes + 255) & ~(size_t)255;
    return p;
  };
  unsigned short* Wk1t  = (unsigned short*)carve((size_t)D_   * NODE_ * 2);
  unsigned short* Wk2t  = (unsigned short*)carve((size_t)NODE_ * D_   * 2);
  unsigned short* Wv1t  = (unsigned short*)carve((size_t)V_   * NODE_ * 2);
  unsigned short* Wv2t  = (unsigned short*)carve((size_t)NODE_ * V_   * 2);
  unsigned short* Wb1t  = (unsigned short*)carve((size_t)D_   * GATE_ * 2);
  unsigned short* poold = (unsigned short*)carve((size_t)B_ * D_ * 2);
  unsigned short* poolv = (unsigned short*)carve((size_t)B_ * V_ * 2);
  unsigned short* H1    = (unsigned short*)carve((size_t)B_ * NODE_ * 2);
  unsigned short* H2    = (unsigned short*)carve((size_t)B_ * NODE_ * 2);
  unsigned short* kbuf  = (unsigned short*)carve((size_t)B_ * D_ * 2);
  unsigned short* vbuf  = (unsigned short*)carve((size_t)B_ * V_ * 2);
  unsigned short* Gbuf  = (unsigned short*)carve((size_t)B_ * GATE_ * 2);
  float*          beta  = (float*)carve((size_t)B_ * 4);

  const float BN_SCALE = 1.0f / sqrtf(1.0f + 1e-5f);

  // ---- 1) convert + transpose weights to bf16 [N,K] ----
  auto cvt = [&](const float* W, unsigned short* Wt, int K, int N) {
    int n = K * N;
    convert_transpose<<<(n + 255) / 256, 256, 0, stream>>>(W, Wt, K, N);
  };
  cvt(Wk1, Wk1t, D_,   NODE_);
  cvt(Wk2, Wk2t, NODE_, D_);
  cvt(Wv1, Wv1t, V_,   NODE_);
  cvt(Wv2, Wv2t, NODE_, V_);
  cvt(Wb1, Wb1t, D_,   GATE_);

  // ---- 2) pooling (one HBM pass over x) ----
  pool_kernel<<<B_, V_, 0, stream>>>(x0, poold, poolv);

  // ---- 3) WMMA GEMM chain ----
  auto gemm_grid = [](int M, int N) { return dim3((unsigned)((M / 16) * (N / 16) / 4)); };
  // k branch
  wmma_gemm<1><<<gemm_grid(B_, NODE_), 128, 0, stream>>>(poold, Wk1t, bk1, H1,
                                                         B_, NODE_, D_, BN_SCALE);
  wmma_gemm<0><<<gemm_grid(B_, D_),    128, 0, stream>>>(H1, Wk2t, bk2, kbuf,
                                                         B_, D_, NODE_, 1.0f);
  normalize_k<<<B_, 256, 0, stream>>>(kbuf);
  // v branch
  wmma_gemm<1><<<gemm_grid(B_, NODE_), 128, 0, stream>>>(poolv, Wv1t, bv1, H2,
                                                         B_, NODE_, V_, BN_SCALE);
  wmma_gemm<0><<<gemm_grid(B_, V_),    128, 0, stream>>>(H2, Wv2t, bv2, vbuf,
                                                         B_, V_, NODE_, 1.0f);
  // gate branch
  wmma_gemm<2><<<gemm_grid(B_, GATE_), 128, 0, stream>>>(poold, Wb1t, bb1, Gbuf,
                                                         B_, GATE_, D_, 1.0f);
  beta_kernel<<<B_ / 4, 128, 0, stream>>>(Gbuf, Wb2, bb2, beta);

  // ---- 4) fused delta update (second HBM pass over x, staged in LDS) ----
  const size_t FUSE_SMEM = (size_t)D_ * V_ * 2 + (size_t)D_ * 2;  // 263168 B
  fuse_kernel<<<B_, V_, FUSE_SMEM, stream>>>(x0, kbuf, vbuf, beta, out);
}
